// QuantUNetFP4_29721173688958
// MI455X (gfx1250) — matrix operand order repaired and checked by reading, not compiled, not run on hardware
//
#include <hip/hip_runtime.h>

// ---------------------------------------------------------------------------
// Types for CDNA5 WMMA (wave32): v_wmma_f32_16x16x32_bf16
// ---------------------------------------------------------------------------
typedef __bf16 v16bf __attribute__((ext_vector_type(16)));
typedef float  v8f   __attribute__((ext_vector_type(8)));

union ABf16 { v16bf v; unsigned u[8]; };

// ---------------------------------------------------------------------------
// bf16 <-> f32 via bit manipulation (round-to-nearest-even); storage = u16
// ---------------------------------------------------------------------------
__device__ __forceinline__ unsigned short f2bf(float f) {
  unsigned x = __builtin_bit_cast(unsigned, f);
  unsigned r = (x + 0x7fffu + ((x >> 16) & 1u)) >> 16;
  return (unsigned short)r;
}
__device__ __forceinline__ float bf2f(unsigned short s) {
  unsigned x = ((unsigned)s) << 16;
  return __builtin_bit_cast(float, x);
}
// power-of-two scale for 4-bit fixed point: 2^(ceil(log2(th)) - 3)
__device__ __forceinline__ float po2_scale(float th) {
  return exp2f(ceilf(log2f(fmaxf(th, 1e-8f))) - 3.0f);
}

// ---------------------------------------------------------------------------
// Implicit-GEMM patch gather (bf16 bits). MODE: 0 = 3x3 pad1, 1 = 1x1,
// 2 = transposed-conv 2x2 stride 2 (input spatial = H/2 x W/2, one tap/pixel).
// Dual input pointers implement channel-concat without materializing it.
// ---------------------------------------------------------------------------
template <int MODE>
__device__ __forceinline__ unsigned fetch_bits(
    const unsigned short* __restrict__ in1, const unsigned short* __restrict__ in2,
    int C1, int C2, int n, int H, int W, int h, int w, int k, int Ktot) {
  if (k >= Ktot) return 0u;
  int ci, hh, ww, Hs = H, Ws = W;
  if (MODE == 0) {
    ci = k / 9; int r = k - ci * 9;               // constant divisor -> mul/shift
    hh = h + r / 3 - 1; ww = w + (r % 3) - 1;
    if ((unsigned)hh >= (unsigned)H || (unsigned)ww >= (unsigned)W) return 0u;
  } else if (MODE == 1) {
    ci = k; hh = h; ww = w;
  } else {
    ci = k >> 2; int t = k & 3;
    if (t != (((h & 1) << 1) | (w & 1))) return 0u;
    hh = h >> 1; ww = w >> 1; Hs = H >> 1; Ws = W >> 1;
  }
  const unsigned short* src; int cc, Cs;
  if (ci < C1) { src = in1; cc = ci; Cs = C1; }
  else         { src = in2; cc = ci - C1; Cs = C2; }
  return (unsigned)src[(((size_t)n * Cs + cc) * Hs + hh) * Ws + ww];
}

// ---------------------------------------------------------------------------
// Core conv kernel: one block = 32 output pixels (two 16-pixel M-tiles, each
// within one W-row since 16 | W). 8 waves x 4 slots cover (mtile, ctile) pairs
// bijectively: mtile = wave&1, ctile = (wave>>1) + 4*s.  A-panels (2 x 16x32
// bf16, ISA 16-bit A layout) are double-buffered in LDS: one barrier per
// K-chunk, next chunk's im2col staged while current chunk's WMMAs issue.
// B panels pre-swizzled per-lane in bpack. f32 accumulation in v8f.
// ---------------------------------------------------------------------------
template <int MODE>
__global__ __launch_bounds__(256) void k_conv_wmma(
    const unsigned short* __restrict__ in1, const unsigned short* __restrict__ in2,
    int C1, int C2, const unsigned* __restrict__ bpack, const float* __restrict__ bias,
    float* __restrict__ out_f, unsigned short* __restrict__ out_b,
    int N, int H, int W, int Cout) {
  __shared__ unsigned a_sh[2][512];  // [buf][mtile*256 + m*16 + kpair_dword]

  const int tid  = threadIdx.x;
  const int lane = tid & 31;
  const int wave = tid >> 5;
  const int p0   = blockIdx.x << 5;
  const int HW   = H * W;

  const int Cin   = C1 + C2;
  constexpr int Ktaps = (MODE == 0) ? 9 : ((MODE == 1) ? 1 : 4);
  const int Ktot  = Cin * Ktaps;
  const int KC    = (Ktot + 31) >> 5;
  const int tiles = Cout >> 4;

  // Hoisted staging coordinates: this thread stages dwords {tid, tid+256}.
  int sn[2], sh2[2], sw2[2], sd2[2];
#pragma unroll
  for (int j = 0; j < 2; ++j) {
    const int dword = tid + (j << 8);
    const int m32 = dword >> 4;          // pixel 0..31 within block
    sd2[j] = dword & 15;                 // K-pair dword 0..15
    const int p = p0 + m32;
    const int n = p / HW; const int r = p - n * HW;
    const int h = r / W;
    sn[j] = n; sh2[j] = h; sw2[j] = r - h * W;
  }

  const int mtile = wave & 1;
  const int cbase = wave >> 1;           // ctile for slot s = cbase + 4*s
  bool has[4];
#pragma unroll
  for (int s = 0; s < 4; ++s) has[s] = (cbase + 4 * s) < tiles;

  v8f acc[4] = {{}, {}, {}, {}};

  const int mA    = lane & 15;
  const int halfA = lane >> 4;
  const int abase = (mtile << 8) + (mA << 4);

  auto stage = [&](int kc) {
#pragma unroll
    for (int j = 0; j < 2; ++j) {
      const int k0 = (kc << 5) + (sd2[j] << 1);
      unsigned lo = fetch_bits<MODE>(in1, in2, C1, C2, sn[j], H, W, sh2[j], sw2[j], k0,     Ktot);
      unsigned hi = fetch_bits<MODE>(in1, in2, C1, C2, sn[j], H, W, sh2[j], sw2[j], k0 + 1, Ktot);
      a_sh[kc & 1][tid + (j << 8)] = lo | (hi << 16);
    }
  };

  stage(0);
  for (int kc = 0; kc < KC; ++kc) {
    __syncthreads();                     // staged panel kc is ready
    if (kc + 1 < KC) stage(kc + 1);      // overlap next im2col with WMMA

    const unsigned* ab = a_sh[kc & 1];
    // ISA 16-bit A 16x32 layout: VGPR v K-pair start = (v/4)*16 + half*8 + (v%4)*2
    ABf16 A;
#pragma unroll
    for (int v = 0; v < 8; ++v) {
      const int d = ((v >> 2) << 3) + (halfA << 2) + (v & 3);
      A.u[v] = ab[abase + d];
    }

#pragma unroll
    for (int s = 0; s < 4; ++s) {
      if (!has[s]) continue;             // wave-uniform: EXEC stays all-ones
      const size_t ct = (size_t)(cbase + 4 * s);
      ABf16 B;
#pragma unroll
      for (int v = 0; v < 8; ++v)
        B.u[v] = bpack[(((ct * KC + kc) << 3) + v) * 32 + lane];
      acc[s] = __builtin_amdgcn_wmma_f32_16x16x32_bf16(false, A.v, false, B.v,
                                                       (short)0, acc[s], false, false);
    }
    if (kc + 1 < KC && has[0])
      __builtin_prefetch((const void*)&bpack[((((size_t)cbase * KC) + kc + 1) << 3) * 32 + lane], 0, 0);
  }

  // Epilogue: D layout -> lane holds col N = lane%16, rows M = (lane/16)*8 + v.
  // Rows map to consecutive W -> contiguous memory: use b128 stores.
  const int pm = p0 + (mtile << 4);
  const int n  = pm / HW; const int r = pm - n * HW;
  const int h  = r / W; const int w0 = r - h * W;
  const int ncol  = lane & 15;
  const int mbase = (lane >> 4) << 3;
#pragma unroll
  for (int s = 0; s < 4; ++s) {
    if (!has[s]) continue;
    const int co = ((cbase + 4 * s) << 4) + ncol;
    const float bv = bias[co];
    const size_t base = (((size_t)n * Cout + co) * H + h) * W + (w0 + mbase);
    float o[8];
#pragma unroll
    for (int v = 0; v < 8; ++v) o[v] = acc[s][v] + bv;
    if (out_f) {
      *(float4*)(out_f + base)     = make_float4(o[0], o[1], o[2], o[3]);
      *(float4*)(out_f + base + 4) = make_float4(o[4], o[5], o[6], o[7]);
    } else {
      uint4 u;
      u.x = (unsigned)f2bf(o[0]) | ((unsigned)f2bf(o[1]) << 16);
      u.y = (unsigned)f2bf(o[2]) | ((unsigned)f2bf(o[3]) << 16);
      u.z = (unsigned)f2bf(o[4]) | ((unsigned)f2bf(o[5]) << 16);
      u.w = (unsigned)f2bf(o[6]) | ((unsigned)f2bf(o[7]) << 16);
      *(uint4*)(out_b + base) = u;
    }
  }
}

// ---------------------------------------------------------------------------
// Weight quantize (int4 fixed-point, exact in bf16) + swizzle into the
// per-lane B-matrix layout: bpack[((tile*KC + kc)*8 + v)*32 + lane],
// lane -> (N = lane%16, Khalf = lane/16), VGPR v -> K = kc*32 + (v/4)*16 + Khalf*8 + 2*(v%4).
// ---------------------------------------------------------------------------
__global__ void k_pack_w(const float* __restrict__ w, const float* __restrict__ slot,
                         unsigned* __restrict__ bpack, int Cin, int Cout, int mode) {
  const int Ktaps = (mode == 0) ? 9 : ((mode == 1) ? 1 : 4);
  const int Ktot  = Cin * Ktaps;
  const int KC    = (Ktot + 31) >> 5;
  const long total = (long)(Cout >> 4) * KC * 256;
  long i = (long)blockIdx.x * 256 + threadIdx.x;
  if (i >= total) return;
  const int lane = (int)(i & 31);
  long j = i >> 5;
  const int v  = (int)(j & 7); j >>= 3;
  const int kc = (int)(j % KC);
  const int tile = (int)(j / KC);

  const float s = po2_scale(slot[0]);
  const int ncol = lane & 15, half = lane >> 4;
  const int co = (tile << 4) + ncol;
  const int k0 = (kc << 5) + ((v >> 2) << 4) + (half << 3) + ((v & 3) << 1);
  unsigned out = 0;
  for (int e = 0; e < 2; ++e) {
    const int k = k0 + e;
    unsigned short bits = 0;
    if (k < Ktot) {
      float wv;
      if (mode == 0)      { int ci = k / 9; int r = k - ci * 9; wv = w[((size_t)co * Cin + ci) * 9 + r]; }
      else if (mode == 1) { wv = w[(size_t)co * Cin + k]; }
      else                { int ci = k >> 2; int t = k & 3; wv = w[(((size_t)ci * Cout + co) << 2) + t]; }
      float q = rintf(fminf(fmaxf(wv / s, -7.f), 7.f)) * s;  // wq: clip->round->rescale
      bits = f2bf(q);
    }
    out |= ((unsigned)bits) << (16 * e);
  }
  bpack[i] = out;
}

// ---------------------------------------------------------------------------
// Elementwise / reduction helpers
// ---------------------------------------------------------------------------
__global__ void k_zero_f32(float* p, int n) {
  int i = blockIdx.x * blockDim.x + threadIdx.x;
  if (i < n) p[i] = 0.f;
}

__global__ void k_f2b(const float* __restrict__ in, unsigned short* __restrict__ out, long n) {
  long i = (long)blockIdx.x * 256 + threadIdx.x;
  if (i < n) out[i] = f2bf(in[i]);
}

__global__ void k_absmax_f32(const float* __restrict__ x, long n, float* slot) {
  float m = 0.f;
  for (long i = (long)blockIdx.x * 256 + threadIdx.x; i < n; i += (long)gridDim.x * 256)
    m = fmaxf(m, fabsf(x[i]));
  __shared__ float sh[256];
  sh[threadIdx.x] = m; __syncthreads();
  for (int st = 128; st > 0; st >>= 1) {
    if (threadIdx.x < st) sh[threadIdx.x] = fmaxf(sh[threadIdx.x], sh[threadIdx.x + st]);
    __syncthreads();
  }
  if (threadIdx.x == 0)
    atomicMax((unsigned*)slot, __builtin_bit_cast(unsigned, sh[0]));
}

__global__ void k_absmax_bf16(const unsigned short* __restrict__ x, long n, float* slot) {
  float m = 0.f;
  for (long i = (long)blockIdx.x * 256 + threadIdx.x; i < n; i += (long)gridDim.x * 256)
    m = fmaxf(m, fabsf(bf2f(x[i])));
  __shared__ float sh[256];
  sh[threadIdx.x] = m; __syncthreads();
  for (int st = 128; st > 0; st >>= 1) {
    if (threadIdx.x < st) sh[threadIdx.x] = fmaxf(sh[threadIdx.x], sh[threadIdx.x + st]);
    __syncthreads();
  }
  if (threadIdx.x == 0)
    atomicMax((unsigned*)slot, __builtin_bit_cast(unsigned, sh[0]));
}

// aq: per-tensor signed int4 activation fake-quant (exact in bf16)
__global__ void k_aq(const unsigned short* __restrict__ in, unsigned short* __restrict__ out,
                     long n, const float* __restrict__ slot) {
  long i = (long)blockIdx.x * 256 + threadIdx.x;
  if (i >= n) return;
  const float s = po2_scale(slot[0]);
  float y = fminf(fmaxf(bf2f(in[i]) / s, -8.f), 7.f);
  out[i] = f2bf(rintf(y) * s);
}

// GroupNorm stats: one block per (n, group); population var; groups = 8.
__global__ void k_gn_stats(const float* __restrict__ x, float* __restrict__ mv,
                           int N, int C, int H, int W) {
  const int ng = blockIdx.x;
  const int n = ng >> 3, g = ng & 7;
  const int cpg = C >> 3;
  const long cnt = (long)cpg * H * W;
  const float* base = x + ((size_t)n * C + (size_t)g * cpg) * H * W;
  float s = 0.f, q = 0.f;
  for (long i = threadIdx.x; i < cnt; i += 256) { float v = base[i]; s += v; q += v * v; }
  __shared__ float shs[256], shq[256];
  shs[threadIdx.x] = s; shq[threadIdx.x] = q; __syncthreads();
  for (int st = 128; st > 0; st >>= 1) {
    if (threadIdx.x < st) { shs[threadIdx.x] += shs[threadIdx.x + st]; shq[threadIdx.x] += shq[threadIdx.x + st]; }
    __syncthreads();
  }
  if (threadIdx.x == 0) {
    float mean = shs[0] / (float)cnt;
    float var  = shq[0] / (float)cnt - mean * mean;
    mv[2 * ng]     = mean;
    mv[2 * ng + 1] = rsqrtf(fmaxf(var, 0.f) + 1e-5f);
  }
}

// Fused GroupNorm-affine + ReLU, f32 in -> bf16 out.
__global__ void k_gn_apply(const float* __restrict__ x, const float* __restrict__ mv,
                           const float* __restrict__ gam, const float* __restrict__ bet,
                           unsigned short* __restrict__ out, int N, int C, int H, int W) {
  long total = (long)N * C * H * W;
  long i = (long)blockIdx.x * 256 + threadIdx.x;
  if (i >= total) return;
  const long hw = (long)H * W;
  int c = (int)((i / hw) % C);
  int n = (int)(i / (hw * C));
  int cpg = C >> 3;
  int ng = (n << 3) + c / cpg;
  float v = (x[i] - mv[2 * ng]) * mv[2 * ng + 1] * gam[c] + bet[c];
  out[i] = f2bf(fmaxf(v, 0.f));
}

__global__ void k_maxpool(const unsigned short* __restrict__ in, unsigned short* __restrict__ out,
                          int N, int C, int H, int W) {
  const int Ho = H >> 1, Wo = W >> 1;
  long total = (long)N * C * Ho * Wo;
  long i = (long)blockIdx.x * 256 + threadIdx.x;
  if (i >= total) return;
  int wo = (int)(i % Wo); long j = i / Wo;
  int ho = (int)(j % Ho); long nc = j / Ho;
  const unsigned short* b = in + (size_t)nc * H * W;
  float m =        bf2f(b[(2 * ho) * W + 2 * wo]);
  m = fmaxf(m, bf2f(b[(2 * ho) * W + 2 * wo + 1]));
  m = fmaxf(m, bf2f(b[(2 * ho + 1) * W + 2 * wo]));
  m = fmaxf(m, bf2f(b[(2 * ho + 1) * W + 2 * wo + 1]));
  out[i] = f2bf(m);
}

// ---------------------------------------------------------------------------
// Host orchestration (all on `stream`; ws bump-allocated; graph-capture safe)
// ---------------------------------------------------------------------------
extern "C" void kernel_launch(void* const* d_in, const int* in_sizes, int n_in,
                              void* d_out, int out_size, void* d_ws, size_t ws_size,
                              hipStream_t stream) {
  (void)in_sizes; (void)n_in; (void)out_size; (void)ws_size;
  const int N = 8;

  size_t off = 0;
  auto alloc = [&](size_t bytes) -> void* {
    off = (off + 255) & ~(size_t)255;
    void* p = (char*)d_ws + off;
    off += bytes;
    return p;
  };
  auto u16buf = [&](long elems) { return (unsigned short*)alloc((size_t)elems * 2); };

  float*    slot = (float*)alloc(sizeof(float));
  float*    mv   = (float*)alloc(64 * 2 * sizeof(float));
  unsigned* pack = (unsigned*)alloc((size_t)524288 * 4);   // max 294912 dwords needed
  float*    tmpf = (float*)alloc((size_t)8388608 * 4);     // largest f32 conv output
  unsigned short* tmpb = u16buf(8388608);
  unsigned short* xb   = u16buf(1572864);
  unsigned short* e1   = u16buf(8388608);
  unsigned short* p1   = u16buf(2097152);
  unsigned short* e2   = u16buf(4194304);
  unsigned short* p2   = u16buf(1048576);
  unsigned short* e3   = u16buf(2097152);
  unsigned short* p3   = u16buf(524288);
  unsigned short* e4   = u16buf(1048576);
  unsigned short* p4   = u16buf(262144);
  unsigned short* bnb  = u16buf(524288);
  unsigned short* upb  = u16buf(8388608);
  unsigned short* d4b  = u16buf(1048576);
  unsigned short* d3b  = u16buf(2097152);
  unsigned short* d2b  = u16buf(4194304);
  unsigned short* d1b  = u16buf(8388608);

  auto F = [&](int i) { return (const float*)d_in[i]; };
  struct BP { const float *w1, *b1, *g1, *be1, *w2, *b2, *g2, *be2; };
  auto getBP = [&](int b) {
    BP p{F(b), F(b + 1), F(b + 2), F(b + 3), F(b + 4), F(b + 5), F(b + 6), F(b + 7)};
    return p;
  };
  auto blocksFor = [](long n) -> unsigned { return (unsigned)((n + 255) / 256); };
  auto capped = [](long n) -> unsigned {
    long b = (n + 255) / 256; return (unsigned)(b < 1024 ? b : 1024);
  };

  { // network input -> bf16
    long n = (long)N * 3 * 256 * 256;
    k_f2b<<<blocksFor(n), 256, 0, stream>>>(F(0), xb, n);
  }

  auto conv = [&](const unsigned short* i1, const unsigned short* i2, int C1, int C2,
                  const float* w, long wcnt, const float* b, int Cout, int H, int W, int mode,
                  float* of, unsigned short* ob) {
    k_zero_f32<<<1, 32, 0, stream>>>(slot, 1);
    k_absmax_f32<<<capped(wcnt), 256, 0, stream>>>(w, wcnt, slot);
    int Ktaps = (mode == 0) ? 9 : ((mode == 1) ? 1 : 4);
    int KC = ((C1 + C2) * Ktaps + 31) >> 5;
    long tot = (long)(Cout >> 4) * KC * 256;
    k_pack_w<<<blocksFor(tot), 256, 0, stream>>>(w, slot, pack, C1 + C2, Cout, mode);
    dim3 grid((unsigned)((long)N * H * W / 32));
    if (mode == 0)
      k_conv_wmma<0><<<grid, 256, 0, stream>>>(i1, i2, C1, C2, pack, b, of, ob, N, H, W, Cout);
    else if (mode == 1)
      k_conv_wmma<1><<<grid, 256, 0, stream>>>(i1, i2, C1, C2, pack, b, of, ob, N, H, W, Cout);
    else
      k_conv_wmma<2><<<grid, 256, 0, stream>>>(i1, i2, C1, C2, pack, b, of, ob, N, H, W, Cout);
  };

  auto gn = [&](const float* xin, const float* g, const float* be, unsigned short* o,
                int C, int H, int W) {
    k_gn_stats<<<N * 8, 256, 0, stream>>>(xin, mv, N, C, H, W);
    long tot = (long)N * C * H * W;
    k_gn_apply<<<blocksFor(tot), 256, 0, stream>>>(xin, mv, g, be, o, N, C, H, W);
  };

  auto aq = [&](unsigned short* buf, long n) {
    k_zero_f32<<<1, 32, 0, stream>>>(slot, 1);
    k_absmax_bf16<<<capped(n), 256, 0, stream>>>(buf, n, slot);
    k_aq<<<blocksFor(n), 256, 0, stream>>>(buf, buf, n, slot);
  };

  auto block = [&](const unsigned short* i1, const unsigned short* i2, int C1, int C2,
                   int Cout, int H, int W, BP P, unsigned short* outb) {
    conv(i1, i2, C1, C2, P.w1, (long)(C1 + C2) * Cout * 9, P.b1, Cout, H, W, 0, tmpf, nullptr);
    gn(tmpf, P.g1, P.be1, tmpb, Cout, H, W);
    conv(tmpb, nullptr, Cout, 0, P.w2, (long)Cout * Cout * 9, P.b2, Cout, H, W, 0, tmpf, nullptr);
    gn(tmpf, P.g2, P.be2, outb, Cout, H, W);
    aq(outb, (long)N * Cout * H * W);
  };

  auto pool = [&](const unsigned short* in, unsigned short* out, int C, int H, int W) {
    long tot = (long)N * C * (H >> 1) * (W >> 1);
    k_maxpool<<<blocksFor(tot), 256, 0, stream>>>(in, out, N, C, H, W);
  };

  // ---- encoder ----
  block(xb, nullptr, 3, 0, 16, 256, 256, getBP(1), e1);
  pool(e1, p1, 16, 256, 256);
  block(p1, nullptr, 16, 0, 32, 128, 128, getBP(9), e2);
  pool(e2, p2, 32, 128, 128);
  block(p2, nullptr, 32, 0, 64, 64, 64, getBP(17), e3);
  pool(e3, p3, 64, 64, 64);
  block(p3, nullptr, 64, 0, 128, 32, 32, getBP(25), e4);
  pool(e4, p4, 128, 32, 32);
  block(p4, nullptr, 128, 0, 256, 16, 16, getBP(33), bnb);

  // ---- decoder (upconv mode=2, skip-concat via dual-input conv) ----
  conv(bnb, nullptr, 256, 0, F(41), (long)256 * 128 * 4, F(42), 128, 32, 32, 2, nullptr, upb);
  block(e4, upb, 128, 128, 128, 32, 32, getBP(43), d4b);
  conv(d4b, nullptr, 128, 0, F(51), (long)128 * 64 * 4, F(52), 64, 64, 64, 2, nullptr, upb);
  block(e3, upb, 64, 64, 64, 64, 64, getBP(53), d3b);
  conv(d3b, nullptr, 64, 0, F(61), (long)64 * 32 * 4, F(62), 32, 128, 128, 2, nullptr, upb);
  block(e2, upb, 32, 32, 32, 128, 128, getBP(63), d2b);
  conv(d2b, nullptr, 32, 0, F(71), (long)32 * 16 * 4, F(72), 16, 256, 256, 2, nullptr, upb);
  block(e1, upb, 16, 16, 16, 256, 256, getBP(73), d1b);

  // ---- final 1x1 conv, f32 output straight to d_out ----
  conv(d1b, nullptr, 16, 0, F(81), (long)80 * 16, F(82), 80, 256, 256, 1, (float*)d_out, nullptr);
}
